// TrfEdgeNetC_79645873537756
// MI455X (gfx1250) — matrix operand order repaired and checked by reading, not compiled, not run on hardware
//
#include <hip/hip_runtime.h>
#include <hip/hip_bf16.h>
#include <stdint.h>

// ---------------------------------------------------------------------------
// Types for CDNA5 WMMA (wave32): 16x16x32 bf16 -> f32 accumulate
// ---------------------------------------------------------------------------
typedef __attribute__((ext_vector_type(16))) __bf16 bf16x16;
typedef __attribute__((ext_vector_type(8)))  float  f32x8;

#define N_NODES 50000
#define N_EDGES 800000

__device__ __forceinline__ unsigned short f2bf(float x) {
  unsigned u = __float_as_uint(x);
  u += 0x7FFFu + ((u >> 16) & 1u);   // round to nearest even
  return (unsigned short)(u >> 16);
}
// monotone float<->uint encoding for atomic max on floats
__device__ __forceinline__ unsigned encf(float f) {
  unsigned b = __float_as_uint(f);
  return b ^ ((b & 0x80000000u) ? 0xFFFFFFFFu : 0x80000000u);
}
__device__ __forceinline__ float decf(unsigned u) {
  unsigned b = (u & 0x80000000u) ? (u ^ 0x80000000u) : ~u;
  return __uint_as_float(b);
}

// ---------------------------------------------------------------------------
// f32 -> bf16 conversion, 4 elems/thread (optionally fused ReLU)
// n must be a multiple of 4 (always true here: N*128 / N*256)
// ---------------------------------------------------------------------------
__global__ void cvt_bf16_kernel(const float* __restrict__ X,
                                unsigned short* __restrict__ Y, int n4, int relu) {
  int i = blockIdx.x * blockDim.x + threadIdx.x;
  if (i >= n4) return;
  float4 v = *(const float4*)(X + (size_t)i * 4);
  if (relu) {
    v.x = fmaxf(v.x, 0.0f); v.y = fmaxf(v.y, 0.0f);
    v.z = fmaxf(v.z, 0.0f); v.w = fmaxf(v.w, 0.0f);
  }
  ushort4 o;
  o.x = f2bf(v.x); o.y = f2bf(v.y); o.z = f2bf(v.z); o.w = f2bf(v.w);
  *(ushort4*)(Y + (size_t)i * 4) = o;
}

// ---------------------------------------------------------------------------
// Pack W[K,M] (row-major f32) into per-lane WMMA B-operand layout (bf16):
//   linear idx = ((mt*kTiles + kt)*32 + lane)*16 + j
//   element    = W[kt*32 + 16*(lane/16) + j][mt*16 + lane%16]   (0 if m >= M)
// ---------------------------------------------------------------------------
__global__ void pack_weights_kernel(const float* __restrict__ W,
                                    unsigned short* __restrict__ Bpack,
                                    int K, int M, int Mpad) {
  int tid = blockIdx.x * blockDim.x + threadIdx.x;
  int total = Mpad * K;
  if (tid >= total) return;
  int kTiles = K >> 5;
  int j    = tid & 15;
  int lane = (tid >> 4) & 31;
  int rest = tid >> 9;
  int kt   = rest % kTiles;
  int mt   = rest / kTiles;
  int k = kt * 32 + 16 * (lane >> 4) + j;
  int m = mt * 16 + (lane & 15);
  float val = (m < M) ? W[(size_t)k * M + m] : 0.0f;
  Bpack[tid] = f2bf(val);
}

// ---------------------------------------------------------------------------
// WMMA GEMM: C[rows, ldOut] = A[rows,K](bf16) @ Wpacked + bias
// one wave -> one 16x16 tile; 8 waves per block; rows % 16 == 0, K % 32 == 0
// ---------------------------------------------------------------------------
__global__ void wmma_gemm_kernel(const unsigned short* __restrict__ Abf,
                                 const unsigned short* __restrict__ Bpack,
                                 const float* __restrict__ bias,
                                 float* __restrict__ C,
                                 int K, int Mpad, int outCols, int ldOut) {
  const int lane = threadIdx.x & 31;
  const int wave = threadIdx.x >> 5;
  const int rowTile = blockIdx.x;
  const int colTile = blockIdx.y * 8 + wave;
  if (colTile * 16 >= Mpad) return;          // wave-uniform: EXEC stays all-ones

  const int g  = lane >> 4;                  // half-wave group
  const int lm = lane & 15;
  union { bf16x16 v; uint4 q[2]; } afrag, bfrag;
  union { f32x8 v; float f[8]; } acc;
#pragma unroll
  for (int r = 0; r < 8; ++r) acc.f[r] = 0.0f;

  const unsigned short* arow = Abf + (size_t)(rowTile * 16 + lm) * K;
  const int kTiles = K >> 5;
  const unsigned short* bbase = Bpack + ((size_t)colTile * kTiles * 32 + lane) * 16;

  for (int kt = 0; kt < kTiles; ++kt) {
    // A: row lm, K runs [kt*32 + 16g .. +15]
    afrag.q[0] = *(const uint4*)(arow + kt * 32 + 16 * g);
    afrag.q[1] = *(const uint4*)(arow + kt * 32 + 16 * g + 8);
    const uint4* bp = (const uint4*)(bbase + (size_t)kt * 512);
    bfrag.q[0] = bp[0];
    bfrag.q[1] = bp[1];
    acc.v = __builtin_amdgcn_wmma_f32_16x16x32_bf16(
        false, afrag.v, false, bfrag.v, (short)0, acc.v, false, false);
  }

  const int col = colTile * 16 + lm;
  if (col < outCols) {
    const float b = bias ? bias[col] : 0.0f;
#pragma unroll
    for (int r = 0; r < 8; ++r) {            // C layout: row = r + 8g, col = lane%16
      const int row = rowTile * 16 + r + 8 * g;
      C[(size_t)row * ldOut + col] = acc.f[r] + b;
    }
  }
}

// ---------------------------------------------------------------------------
// Segment-softmax state init
// ---------------------------------------------------------------------------
__global__ void init_seg_kernel(unsigned* __restrict__ mmax,
                                float* __restrict__ denom, int n) {
  int i = blockIdx.x * blockDim.x + threadIdx.x;
  if (i >= n) return;
  mmax[i] = 0x007FFFFFu;  // encf(-inf)
  denom[i] = 0.0f;
}

// ---------------------------------------------------------------------------
// Edge pass 1: alpha[e,h] = (q[dst] . (k[src] + ea@We)) / sqrt(C); seg-max.
// One wave per edge; lane owns DOUT/32 channels (float4-vectorized); head = lane/8.
// ---------------------------------------------------------------------------
template <int DOUT>
__global__ void edge_alpha_kernel(const float* __restrict__ q, const float* __restrict__ k,
                                  const float* __restrict__ ea, const float* __restrict__ We,
                                  const int* __restrict__ src, const int* __restrict__ dst,
                                  float* __restrict__ alpha, unsigned* __restrict__ mmax,
                                  int E) {
  const int PC = DOUT / 32;                  // 4 or 8 channels per lane
  const int lane = threadIdx.x & 31;
  const int e = blockIdx.x * (blockDim.x >> 5) + (threadIdx.x >> 5);
  if (e >= E) return;
  const int s = src[e], d = dst[e];

  float eav[16];
#pragma unroll
  for (int t4 = 0; t4 < 4; ++t4) {           // ea row: 4x b128
    float4 r = *(const float4*)(ea + (size_t)e * 16 + t4 * 4);
    eav[t4 * 4 + 0] = r.x; eav[t4 * 4 + 1] = r.y;
    eav[t4 * 4 + 2] = r.z; eav[t4 * 4 + 3] = r.w;
  }

  float dot = 0.0f;
#pragma unroll
  for (int cc = 0; cc < PC / 4; ++cc) {
    const int ch0 = lane * PC + cc * 4;      // 16-B aligned channel group
    float4 ev = make_float4(0.f, 0.f, 0.f, 0.f);
#pragma unroll
    for (int t = 0; t < 16; ++t) {
      const float w = eav[t];
      const float4 wr = *(const float4*)(We + (size_t)t * DOUT + ch0);
      ev.x += w * wr.x; ev.y += w * wr.y; ev.z += w * wr.z; ev.w += w * wr.w;
    }
    const float4 kv = *(const float4*)(k + (size_t)s * DOUT + ch0);
    const float4 qv = *(const float4*)(q + (size_t)d * DOUT + ch0);
    dot += qv.x * (kv.x + ev.x) + qv.y * (kv.y + ev.y) +
           qv.z * (kv.z + ev.z) + qv.w * (kv.w + ev.w);
  }
  dot += __shfl_xor(dot, 1, 32);
  dot += __shfl_xor(dot, 2, 32);
  dot += __shfl_xor(dot, 4, 32);
  if ((lane & 7) == 0) {
    const int h = lane >> 3;
    const float a = dot * (1.0f / sqrtf((float)(DOUT / 4)));
    alpha[(size_t)e * 4 + h] = a;
    atomicMax(&mmax[(size_t)d * 4 + h], encf(a));
  }
}

// ---------------------------------------------------------------------------
// Edge pass 2: p = exp(alpha - max[dst]); denom[dst] += p (per edge,head)
// ---------------------------------------------------------------------------
__global__ void edge_softmax_kernel(const int* __restrict__ dst,
                                    const unsigned* __restrict__ mmax,
                                    float* __restrict__ alpha,
                                    float* __restrict__ denom, int EH) {
  int i = blockIdx.x * blockDim.x + threadIdx.x;
  if (i >= EH) return;
  int e = i >> 2, h = i & 3;
  int d = dst[e];
  float m = decf(mmax[d * 4 + h]);
  float p = __expf(alpha[i] - m);
  alpha[i] = p;
  atomicAdd(&denom[d * 4 + h], p);
}

// ---------------------------------------------------------------------------
// Edge pass 3: out[dst] += (p/denom) * (v[src] + ea@We)
// ---------------------------------------------------------------------------
template <int DOUT>
__global__ void edge_scatter_kernel(const float* __restrict__ v, const float* __restrict__ ea,
                                    const float* __restrict__ We,
                                    const int* __restrict__ src, const int* __restrict__ dst,
                                    const float* __restrict__ alpha,
                                    const float* __restrict__ denom,
                                    float* __restrict__ out, int E) {
  const int PC = DOUT / 32;
  const int lane = threadIdx.x & 31;
  const int e = blockIdx.x * (blockDim.x >> 5) + (threadIdx.x >> 5);
  if (e >= E) return;
  const int s = src[e], d = dst[e];
  const int h = lane >> 3;
  const float a = alpha[(size_t)e * 4 + h] / (denom[(size_t)d * 4 + h] + 1e-16f);

  float eav[16];
#pragma unroll
  for (int t4 = 0; t4 < 4; ++t4) {
    float4 r = *(const float4*)(ea + (size_t)e * 16 + t4 * 4);
    eav[t4 * 4 + 0] = r.x; eav[t4 * 4 + 1] = r.y;
    eav[t4 * 4 + 2] = r.z; eav[t4 * 4 + 3] = r.w;
  }

#pragma unroll
  for (int cc = 0; cc < PC / 4; ++cc) {
    const int ch0 = lane * PC + cc * 4;
    float4 ev = make_float4(0.f, 0.f, 0.f, 0.f);
#pragma unroll
    for (int t = 0; t < 16; ++t) {
      const float w = eav[t];
      const float4 wr = *(const float4*)(We + (size_t)t * DOUT + ch0);
      ev.x += w * wr.x; ev.y += w * wr.y; ev.z += w * wr.z; ev.w += w * wr.w;
    }
    const float4 vv = *(const float4*)(v + (size_t)s * DOUT + ch0);
    float* o = out + (size_t)d * DOUT + ch0;
    atomicAdd(o + 0, a * (vv.x + ev.x));
    atomicAdd(o + 1, a * (vv.y + ev.y));
    atomicAdd(o + 2, a * (vv.z + ev.z));
    atomicAdd(o + 3, a * (vv.w + ev.w));
  }
}

// ---------------------------------------------------------------------------
// Host orchestration
// ---------------------------------------------------------------------------
static inline int ceildiv(int a, int b) { return (a + b - 1) / b; }

extern "C" void kernel_launch(void* const* d_in, const int* in_sizes, int n_in,
                              void* d_out, int out_size, void* d_ws, size_t ws_size,
                              hipStream_t stream) {
  (void)in_sizes; (void)n_in; (void)out_size; (void)ws_size;
  const int N = N_NODES, E = N_EDGES;

  const float* x  = (const float*)d_in[0];
  const int*   ei = (const int*)d_in[1];
  const float* ea = (const float*)d_in[2];
  const int* src = ei;
  const int* dst = ei + E;

  // workspace carve (256-B aligned regions)
  char* p = (char*)d_ws;
  auto carve = [&](size_t bytes) -> char* {
    char* r = p; p += (bytes + 255) & ~(size_t)255; return r;
  };
  unsigned short* hbf   = (unsigned short*)carve((size_t)N * 256 * 2);
  float*          qb    = (float*)carve((size_t)N * 256 * 4);
  float*          kb    = (float*)carve((size_t)N * 256 * 4);
  float*          vb    = (float*)carve((size_t)N * 256 * 4);
  float*          ob    = (float*)carve((size_t)N * 256 * 4);
  float*          alpha = (float*)carve((size_t)E * 4 * 4);
  unsigned*       mmax  = (unsigned*)carve((size_t)N * 4 * 4);
  float*          denom = (float*)carve((size_t)N * 4 * 4);
  unsigned short* packW = (unsigned short*)carve((size_t)256 * 256 * 2);

  auto gemm = [&](const unsigned short* A, const float* W, const float* b, float* C,
                  int K, int M, int Mpad, int outCols, int ldOut) {
    int total = Mpad * K;
    pack_weights_kernel<<<ceildiv(total, 256), 256, 0, stream>>>(W, packW, K, M, Mpad);
    dim3 grid(N / 16, ceildiv(Mpad, 128));
    wmma_gemm_kernel<<<grid, 256, 0, stream>>>(A, packW, b, C, K, Mpad, outCols, ldOut);
  };

  // layer-0 input staging
  cvt_bf16_kernel<<<ceildiv(N * 128 / 4, 256), 256, 0, stream>>>(x, hbf, N * 128 / 4, 0);

  int K = 128;
  for (int li = 0; li < 3; ++li) {
    const int dout = (li == 2) ? 256 : 128;
    const int base = 3 + li * 9;   // Wq,Wk,Wv,Ws,bq,bk,bv,bs,We
    const float* Wq = (const float*)d_in[base + 0];
    const float* Wk = (const float*)d_in[base + 1];
    const float* Wv = (const float*)d_in[base + 2];
    const float* Ws = (const float*)d_in[base + 3];
    const float* bq = (const float*)d_in[base + 4];
    const float* bk = (const float*)d_in[base + 5];
    const float* bv = (const float*)d_in[base + 6];
    const float* bs = (const float*)d_in[base + 7];
    const float* We = (const float*)d_in[base + 8];

    gemm(hbf, Wq, bq, qb, K, dout, dout, dout, dout);
    gemm(hbf, Wk, bk, kb, K, dout, dout, dout, dout);
    gemm(hbf, Wv, bv, vb, K, dout, dout, dout, dout);
    gemm(hbf, Ws, bs, ob, K, dout, dout, dout, dout);   // skip/root into out accumulator

    init_seg_kernel<<<ceildiv(N * 4, 256), 256, 0, stream>>>(mmax, denom, N * 4);

    const int eblocks = ceildiv(E, 8);   // 8 waves (edges) per 256-thread block
    if (dout == 128) {
      edge_alpha_kernel<128><<<eblocks, 256, 0, stream>>>(qb, kb, ea, We, src, dst, alpha, mmax, E);
    } else {
      edge_alpha_kernel<256><<<eblocks, 256, 0, stream>>>(qb, kb, ea, We, src, dst, alpha, mmax, E);
    }
    edge_softmax_kernel<<<ceildiv(E * 4, 256), 256, 0, stream>>>(dst, mmax, alpha, denom, E * 4);
    if (dout == 128) {
      edge_scatter_kernel<128><<<eblocks, 256, 0, stream>>>(vb, ea, We, src, dst, alpha, denom, ob, E);
    } else {
      edge_scatter_kernel<256><<<eblocks, 256, 0, stream>>>(vb, ea, We, src, dst, alpha, denom, ob, E);
    }

    // stage next-layer activations (ReLU after layers 1 and 2 only)
    cvt_bf16_kernel<<<ceildiv(N * dout / 4, 256), 256, 0, stream>>>(ob, hbf, N * dout / 4, (li < 2) ? 1 : 0);
    K = dout;
  }

  // classifier: [N,256] @ [256,10] + bc  (M padded to 16, store-guarded)
  const float* Wc = (const float*)d_in[30];
  const float* bc = (const float*)d_in[31];
  gemm(hbf, Wc, bc, (float*)d_out, 256, 10, 16, 10, 10);
}
